// WeakLayer_2405181686455
// MI455X (gfx1250) — matrix operand order repaired and checked by reading, not compiled, not run on hardware
//
#include <hip/hip_runtime.h>
#include <hip/hip_fp16.h>

#define NPTS 32768
#define NSN  16
#define CH   128
static constexpr float EPSBN = 1e-5f;

typedef __attribute__((ext_vector_type(16))) _Float16 v16h;
typedef __attribute__((ext_vector_type(8)))  _Float16 v8h;
typedef __attribute__((ext_vector_type(8)))  float    v8f;
typedef __attribute__((ext_vector_type(4)))  int      v4i;

#ifndef __has_builtin
#define __has_builtin(x) 0
#endif
#if __has_builtin(__builtin_amdgcn_global_load_async_to_lds_b128) && \
    __has_builtin(__builtin_amdgcn_s_wait_asynccnt)
#define HAVE_ASYNC 1
#else
#define HAVE_ASYNC 0
#endif

#if HAVE_ASYNC
__device__ __forceinline__ void async_cp16(const float* g, float* l) {
  // (global v4i* src, local v4i* dst, imm offset, imm cpol)
  __builtin_amdgcn_global_load_async_to_lds_b128(
      (__attribute__((address_space(1))) v4i*)(g),
      (__attribute__((address_space(3))) v4i*)(l), 0, 0);
}
#endif

__device__ __forceinline__ v16h cat8(v8h lo, v8h hi) {
  return __builtin_shufflevector(lo, hi, 0,1,2,3,4,5,6,7,8,9,10,11,12,13,14,15);
}

// A fragment (16x32 f16, MxK) from an f16 buffer, row stride ldh halves.
// ISA layout: lanes 0-15 m=lane, K {0..7,16..23}; lanes 16-31 K {8..15,24..31}.
__device__ __forceinline__ v16h load_a_h(const _Float16* H, int mbase, int ldh, int kk, int lane) {
  int m = mbase + (lane & 15);
  int koff = (lane >> 4) * 8;
  const v8h* p0 = (const v8h*)(H + (size_t)m * ldh + kk + koff);
  const v8h* p1 = (const v8h*)(H + (size_t)m * ldh + kk + 16 + koff);
  return cat8(p0[0], p1[0]);
}

// Same A fragment built from fp32 row-major data (converted on the fly).
__device__ __forceinline__ v16h load_a_f32(const float* X, int mbase, int ldx, int kk, int lane) {
  int m = mbase + (lane & 15);
  int koff = (lane >> 4) * 8;
  const float* p0 = X + (size_t)m * ldx + kk + koff;
  const float* p1 = X + (size_t)m * ldx + kk + 16 + koff;
  v16h r;
#pragma unroll
  for (int j = 0; j < 8; ++j) { r[j] = (_Float16)p0[j]; r[8 + j] = (_Float16)p1[j]; }
  return r;
}

// B fragment (32x16 KxN) for out = act @ W^T with W row-major (out x in) f16.
// Lanes 0-15 K {kk..kk+7, kk+16..kk+23}, lanes 16-31 K {kk+8..kk+15, kk+24..kk+31}; lane&15 selects column.
__device__ __forceinline__ v16h load_b_h(const _Float16* W, int cbase, int ldw, int kk, int lane) {
  int c = cbase + (lane & 15);
  int koff = (lane >> 4) * 8;
  const v8h* q = (const v8h*)(W + (size_t)c * ldw + kk + koff);
  return cat8(q[0], q[2]);
}

// ---------------- weight fp32 -> fp16 ----------------
__global__ void k_cvt(const float* Wq, const float* Wk, const float* Wv,
                      const float* W1, const float* W2, _Float16* dst) {
  int i = blockIdx.x * 256 + threadIdx.x;
  if (i >= 5 * 16384) return;
  int which = i >> 14, j = i & 16383;
  const float* s = which == 0 ? Wq : which == 1 ? Wk : which == 2 ? Wv : which == 3 ? W1 : W2;
  dst[i] = (_Float16)s[j];
}

// ---------------- q/k/v projections via WMMA ----------------
__global__ void __launch_bounds__(128) k_proj(
    const float* x, const _Float16* Wqh, const _Float16* Wkh, const _Float16* Wvh,
    const float* bq, const float* bk, const float* bv,
    float* xq, float* xk, float* xv) {
  int t = threadIdx.x, lane = t & 31, wv = t >> 5;
  int n0 = blockIdx.x * 16;
  v16h A[4];
#pragma unroll
  for (int k4 = 0; k4 < 4; ++k4) A[k4] = load_a_f32(x, n0, CH, k4 * 32, lane);
  int nlo = lane & 15, mhi = (lane >> 4) * 8;
#pragma unroll
  for (int i = 0; i < 6; ++i) {
    int tt = wv * 6 + i;                 // 24 tiles = 3 weights x 8 col-tiles
    int wm = tt >> 3, ct = tt & 7;
    const _Float16* Wh = wm == 0 ? Wqh : (wm == 1 ? Wkh : Wvh);
    const float* bb = wm == 0 ? bq : (wm == 1 ? bk : bv);
    float* op = wm == 0 ? xq : (wm == 1 ? xk : xv);
    float b0 = bb[ct * 16 + nlo];
    v8f acc;
#pragma unroll
    for (int r = 0; r < 8; ++r) acc[r] = b0;
#pragma unroll
    for (int k4 = 0; k4 < 4; ++k4) {
      v16h B = load_b_h(Wh, ct * 16, CH, k4 * 32, lane);
      acc = __builtin_amdgcn_wmma_f32_16x16x32_f16(false, A[k4], false, B,
                                                   (short)0, acc, false, false);
    }
#pragma unroll
    for (int r = 0; r < 8; ++r) {
      int m = r + mhi;
      op[(size_t)(n0 + m) * CH + ct * 16 + nlo] = acc[r];
    }
  }
}

// ---------------- x_uni ----------------
__global__ void __launch_bounds__(256) k_xu(const float* x, const float* Wu, const float* bu,
                                            const float* gu, const float* bnub, float* xu) {
  __shared__ float wsh[CH];
  int t = threadIdx.x;
  if (t < CH) wsh[t] = Wu[t];
  __syncthreads();
  int n = blockIdx.x * 256 + t;
  const float4* xr4 = (const float4*)(x + (size_t)n * CH);
  const float4* wr4 = (const float4*)wsh;
  float acc = 0.f;
#pragma unroll 4
  for (int i = 0; i < CH / 4; ++i) {
    float4 xv = xr4[i], wv = wr4[i];
    acc += xv.x * wv.x + xv.y * wv.y + xv.z * wv.z + xv.w * wv.w;
  }
  float rs = rsqrtf(1.f + EPSBN);
  float v = (acc + bu[0]) * (gu[0] * rs) + bnub[0];
  xu[n] = v > 0.f ? v : 0.f;
}

__global__ void k_zero(float* p, int n) {
  int i = blockIdx.x * 256 + threadIdx.x;
  if (i < n) p[i] = 0.f;
}

// ---------------- wd partial reduction: wd[c,s] = sum_n (xk[knn]-xq)*xu ----------------
__global__ void __launch_bounds__(128) k_wd(const float* xq, const float* xk, const float* xu,
                                            const int* knn, float* wdraw) {
  int c = threadIdx.x;
  int nb = blockIdx.x * 128;
  float acc[NSN];
#pragma unroll
  for (int s = 0; s < NSN; ++s) acc[s] = 0.f;
  for (int n = nb; n < nb + 128; ++n) {
    float xun = xu[n];
    if (xun == 0.f) continue;          // relu zeros -> uniform skip, halves gather traffic
    float xqc = xq[(size_t)n * CH + c];
#pragma unroll
    for (int s = 0; s < NSN; ++s) {
      int gi = knn[n * NSN + s];
      acc[s] += (xk[(size_t)gi * CH + c] - xqc) * xun;
    }
  }
#pragma unroll
  for (int s = 0; s < NSN; ++s) atomicAdd(&wdraw[c * NSN + s], acc[s]);
}

// softmax over s; also emit wfx[s,c] = a[s] * softmax(wd)[c,s] for the fused kernel
__global__ void k_wdsm(const float* wdraw, const float* aP, float* wdsm, float* wfx) {
  int c = threadIdx.x;  // 128 threads
  float v[NSN], mx = -1e30f;
#pragma unroll
  for (int s = 0; s < NSN; ++s) { v[s] = wdraw[c * NSN + s]; mx = fmaxf(mx, v[s]); }
  float sum = 0.f;
#pragma unroll
  for (int s = 0; s < NSN; ++s) { v[s] = __expf(v[s] - mx); sum += v[s]; }
  float inv = 1.f / sum;
#pragma unroll
  for (int s = 0; s < NSN; ++s) {
    float r = v[s] * inv;
    wdsm[c * NSN + s] = r;
    wfx[s * CH + c] = aP[s] * r;
  }
}

// ---------------- we[n,s] = softmax_s(x @ wd) ----------------
__global__ void __launch_bounds__(256) k_we(const float* x, const float* wdsm, float* we) {
  __shared__ float wsh[CH * NSN];
  int t = threadIdx.x;
  for (int i = t; i < CH * NSN; i += 256) wsh[i] = wdsm[i];
  __syncthreads();
  int n = blockIdx.x * 256 + t;
  const float4* xr4 = (const float4*)(x + (size_t)n * CH);
  float acc[NSN];
#pragma unroll
  for (int s = 0; s < NSN; ++s) acc[s] = 0.f;
  for (int i4 = 0; i4 < CH / 4; ++i4) {
    float4 xv = xr4[i4];
    const float* w0 = &wsh[(i4 * 4) * NSN];
#pragma unroll
    for (int s = 0; s < NSN; ++s)
      acc[s] += xv.x * w0[s] + xv.y * w0[NSN + s] + xv.z * w0[2 * NSN + s] + xv.w * w0[3 * NSN + s];
  }
  float mx = -1e30f;
#pragma unroll
  for (int s = 0; s < NSN; ++s) mx = fmaxf(mx, acc[s]);
  float sum = 0.f;
#pragma unroll
  for (int s = 0; s < NSN; ++s) { acc[s] = __expf(acc[s] - mx); sum += acc[s]; }
  float inv = 1.f / sum;
#pragma unroll
  for (int s = 0; s < NSN; ++s) we[(size_t)n * NSN + s] = acc[s] * inv;
}

// ---------------- fused main kernel: 4 points/block, 4 waves, rows = point*16 + s ----------------
__global__ void __launch_bounds__(128) k_main(
    const float* p, const int* knn,
    const float* Wp1, const float* bp1, const float* gp, const float* bnpb,
    const float* Wp2, const float* bp2,
    const float* Wa1, const float* ba1, const float* ga, const float* bnab,
    const float* Wa2, const float* ba2,
    const float* gw1, const float* bnw1b, const _Float16* W1h, const float* bw1,
    const float* gw2, const float* bnw2b, const _Float16* W2h, const float* bw2,
    const float* bP, const float* dP,
    const float* wfx, const float* weP, const float* xv0,
    float* out) {
  __shared__ __align__(16) _Float16 hbuf[64 * 136];  // f16 activations, padded stride (17408 B)
  __shared__ __align__(16) _Float16 prb[64 * 129];   // p_r in f16 (16512 B)
  __shared__ __align__(16) float wfx_s[NSN * CH];    // a[s]*wf[s,c]
  __shared__ __align__(16) float dP_s[NSN * CH];
  __shared__ __align__(16) float bP_s[NSN * CH];
  __shared__ int idxs[4 * NSN];

  int t = threadIdx.x, lane = t & 31, wv = t >> 5;
  int n0 = blockIdx.x * 4;
  float rs = rsqrtf(1.f + EPSBN);

  // ---- stage (s,c) tables into LDS; async path uses ASYNCcnt-tracked DMA
#if HAVE_ASYNC
#pragma unroll
  for (int i = 0; i < 4; ++i) {
    int off = t * 4 + i * 512;
    async_cp16(wfx + off, wfx_s + off);
    async_cp16(dP + off, dP_s + off);
    async_cp16(bP + off, bP_s + off);
  }
#else
  for (int i = t; i < NSN * CH; i += 128) {
    wfx_s[i] = wfx[i]; dP_s[i] = dP[i]; bP_s[i] = bP[i];
  }
#endif
  if (t < 64) {
    int gi = knn[(size_t)n0 * NSN + t];
    idxs[t] = gi;
    // prefetch the x_v gather rows now; consumed only at the very end of the kernel
    const char* base = (const char*)(xv0 + (size_t)gi * CH);
    __builtin_prefetch(base, 0, 3);
    __builtin_prefetch(base + 128, 0, 3);
    __builtin_prefetch(base + 256, 0, 3);
    __builtin_prefetch(base + 384, 0, 3);
  }
#if HAVE_ASYNC
  __builtin_amdgcn_s_wait_asynccnt(0);
#endif
  __syncthreads();

  // ---- front end: thread -> row t&63, col-half t>>6 (wave-uniform -> scalar param loads)
  {
    int row = t & 63;
    int pt = row >> 4, s = row & 15;
    int gi = idxs[pt * NSN + s];
    int nc = n0 + pt;
    float ex = p[(size_t)gi * 3 + 0] - p[(size_t)nc * 3 + 0];
    float ey = p[(size_t)gi * 3 + 1] - p[(size_t)nc * 3 + 1];
    float ez = p[(size_t)gi * 3 + 2] - p[(size_t)nc * 3 + 2];
    float d3  = sqrtf(ex * ex + ey * ey + ez * ez);
    float dxy = sqrtf(ex * ex + ey * ey);
    float cosT = d3  > 0.f ? dxy / d3  : 0.f;   // nan_to_num(0/0)=0
    float cosA = dxy > 0.f ? ey  / dxy : 0.f;
    float f0 = d3, f1 = cosA, f2 = cosT;
    float t1[3], t2[3];
#pragma unroll
    for (int j = 0; j < 3; ++j) {
      float u = Wa1[j * 3 + 0] * f0 + Wa1[j * 3 + 1] * f1 + Wa1[j * 3 + 2] * f2 + ba1[j];
      u = u * (ga[j] * rs) + bnab[j];
      t1[j] = u > 0.f ? u : 0.f;
      float v = Wp1[j * 3 + 0] * ex + Wp1[j * 3 + 1] * ey + Wp1[j * 3 + 2] * ez + bp1[j];
      v = v * (gp[j] * rs) + bnpb[j];
      t2[j] = v > 0.f ? v : 0.f;
    }
    float wev = weP[(size_t)nc * NSN + s];
    int cub = __builtin_amdgcn_readfirstlane((t >> 6) * 64);  // wave-uniform column base
    for (int cc = 0; cc < 64; ++cc) {
      int c = cub + cc;  // uniform -> Wa2/Wp2/biases come from scalar (constant) loads
      float af = Wa2[c * 3 + 0] * t1[0] + Wa2[c * 3 + 1] * t1[1] + Wa2[c * 3 + 2] * t1[2] + ba2[c];
      float pp = Wp2[c * 3 + 0] * t2[0] + Wp2[c * 3 + 1] * t2[1] + Wp2[c * 3 + 2] * t2[2] + bp2[c];
      float prv = pp + dP_s[s * CH + c] * af;
      prb[row * 129 + c] = (_Float16)prv;
      float w0 = wfx_s[s * CH + c] + bP_s[s * CH + c] * wev + prv;
      float h = w0 * (gw1[c] * rs) + bnw1b[c];
      hbuf[row * 136 + c] = (_Float16)(h > 0.f ? h : 0.f);
    }
  }
  __syncthreads();

  int nlo = lane & 15, mhi = (lane >> 4) * 8;
  v8f acc[8];

  // ---- GEMM1: w1 = relu'd(w0) @ Ww1^T + bw1 (wave's M-tile == its own point)
#pragma unroll
  for (int nt = 0; nt < 8; ++nt) {
    float bb = bw1[nt * 16 + nlo];
#pragma unroll
    for (int r = 0; r < 8; ++r) acc[nt][r] = bb;
  }
#pragma unroll
  for (int k4 = 0; k4 < 4; ++k4) {
    v16h A = load_a_h(hbuf, wv * 16, 136, k4 * 32, lane);
#pragma unroll
    for (int nt = 0; nt < 8; ++nt) {
      v16h B = load_b_h(W1h, nt * 16, CH, k4 * 32, lane);
      acc[nt] = __builtin_amdgcn_wmma_f32_16x16x32_f16(false, A, false, B,
                                                       (short)0, acc[nt], false, false);
    }
  }
  __syncthreads();
  // bn2 + relu -> hbuf (each wave rewrites only its own 16 rows)
#pragma unroll
  for (int nt = 0; nt < 8; ++nt) {
    int c = nt * 16 + nlo;
    float k2 = gw2[c] * rs, b2 = bnw2b[c];
#pragma unroll
    for (int r = 0; r < 8; ++r) {
      float v2 = acc[nt][r] * k2 + b2;
      hbuf[(wv * 16 + r + mhi) * 136 + c] = (_Float16)(v2 > 0.f ? v2 : 0.f);
    }
  }
  __syncthreads();

  // ---- GEMM2: w2 = h2 @ Ww2^T + bw2
#pragma unroll
  for (int nt = 0; nt < 8; ++nt) {
    float bb = bw2[nt * 16 + nlo];
#pragma unroll
    for (int r = 0; r < 8; ++r) acc[nt][r] = bb;
  }
#pragma unroll
  for (int k4 = 0; k4 < 4; ++k4) {
    v16h A = load_a_h(hbuf, wv * 16, 136, k4 * 32, lane);
#pragma unroll
    for (int nt = 0; nt < 8; ++nt) {
      v16h B = load_b_h(W2h, nt * 16, CH, k4 * 32, lane);
      acc[nt] = __builtin_amdgcn_wmma_f32_16x16x32_f16(false, A, false, B,
                                                       (short)0, acc[nt], false, false);
    }
  }

  // ---- softmax over s inside D layout + final gather-reduce to out
#pragma unroll
  for (int nt = 0; nt < 8; ++nt) {
    float mx = acc[nt][0];
#pragma unroll
    for (int r = 1; r < 8; ++r) mx = fmaxf(mx, acc[nt][r]);
    mx = fmaxf(mx, __shfl_xor(mx, 16, 32));       // combine s 0-7 / 8-15 lane halves
    float exv[8], ssum = 0.f;
#pragma unroll
    for (int r = 0; r < 8; ++r) { exv[r] = __expf(acc[nt][r] - mx); ssum += exv[r]; }
    ssum += __shfl_xor(ssum, 16, 32);
    float inv = 1.f / ssum;
    int c = nt * 16 + nlo;
    float o = 0.f;
#pragma unroll
    for (int r = 0; r < 8; ++r) {
      int s = r + mhi;
      float prv = (float)prb[(wv * 16 + s) * 129 + c];
      int gi = idxs[wv * NSN + s];
      float xvv = xv0[(size_t)gi * CH + c];
      o += (xvv + prv) * (exv[r] * inv);
    }
    o += __shfl_xor(o, 16, 32);
    if (lane < 16) out[(size_t)(n0 + wv) * CH + c] = o;
  }
}

extern "C" void kernel_launch(void* const* d_in, const int* in_sizes, int n_in,
                              void* d_out, int out_size, void* d_ws, size_t ws_size,
                              hipStream_t stream) {
  (void)in_sizes; (void)n_in; (void)out_size; (void)ws_size;
  const float* p    = (const float*)d_in[0];
  const float* x    = (const float*)d_in[1];
  const int*   knn  = (const int*)d_in[2];
  const float* Wq   = (const float*)d_in[3];
  const float* bq   = (const float*)d_in[4];
  const float* Wk   = (const float*)d_in[5];
  const float* bk   = (const float*)d_in[6];
  const float* Wv   = (const float*)d_in[7];
  const float* bvv  = (const float*)d_in[8];
  const float* Wu   = (const float*)d_in[9];
  const float* bu   = (const float*)d_in[10];
  const float* gu   = (const float*)d_in[11];
  const float* bnub = (const float*)d_in[12];
  const float* Wp1  = (const float*)d_in[13];
  const float* bp1  = (const float*)d_in[14];
  const float* gp   = (const float*)d_in[15];
  const float* bnpb = (const float*)d_in[16];
  const float* Wp2  = (const float*)d_in[17];
  const float* bp2  = (const float*)d_in[18];
  const float* Wa1  = (const float*)d_in[19];
  const float* ba1  = (const float*)d_in[20];
  const float* ga   = (const float*)d_in[21];
  const float* bnab = (const float*)d_in[22];
  const float* Wa2  = (const float*)d_in[23];
  const float* ba2  = (const float*)d_in[24];
  const float* gw1  = (const float*)d_in[25];
  const float* bnw1b= (const float*)d_in[26];
  const float* Ww1  = (const float*)d_in[27];
  const float* bw1  = (const float*)d_in[28];
  const float* gw2  = (const float*)d_in[29];
  const float* bnw2b= (const float*)d_in[30];
  const float* Ww2  = (const float*)d_in[31];
  const float* bw2  = (const float*)d_in[32];
  const float* aP   = (const float*)d_in[33];
  const float* bP   = (const float*)d_in[34];
  const float* dP   = (const float*)d_in[35];
  float* out = (float*)d_out;

  // Workspace layout (~50.3 MB total)
  float* ws = (float*)d_ws;
  const size_t NC = (size_t)NPTS * CH;
  float* xq    = ws;
  float* xk0   = xq + NC;
  float* xv0   = xk0 + NC;
  float* xu    = xv0 + NC;
  float* wdraw = xu + NPTS;
  float* wdsm  = wdraw + CH * NSN;
  float* wfx   = wdsm + CH * NSN;
  float* weB   = wfx + CH * NSN;
  _Float16* hW = (_Float16*)(weB + (size_t)NPTS * NSN);  // 16B-aligned offset
  _Float16* Wqh = hW;
  _Float16* Wkh = hW + 16384;
  _Float16* Wvh = hW + 32768;
  _Float16* W1h = hW + 49152;
  _Float16* W2h = hW + 65536;

  k_cvt <<<320, 256, 0, stream>>>(Wq, Wk, Wv, Ww1, Ww2, hW);
  k_proj<<<NPTS / 16, 128, 0, stream>>>(x, Wqh, Wkh, Wvh, bq, bk, bvv, xq, xk0, xv0);
  k_xu  <<<NPTS / 256, 256, 0, stream>>>(x, Wu, bu, gu, bnub, xu);
  k_zero<<<(CH * NSN + 255) / 256, 256, 0, stream>>>(wdraw, CH * NSN);
  k_wd  <<<NPTS / 128, 128, 0, stream>>>(xq, xk0, xu, knn, wdraw);
  k_wdsm<<<1, 128, 0, stream>>>(wdraw, aP, wdsm, wfx);
  k_we  <<<NPTS / 256, 256, 0, stream>>>(x, wdsm, weB);
  k_main<<<NPTS / 4, 128, 0, stream>>>(p, knn, Wp1, bp1, gp, bnpb, Wp2, bp2,
                                       Wa1, ba1, ga, bnab, Wa2, ba2,
                                       gw1, bnw1b, W1h, bw1, gw2, bnw2b, W2h, bw2,
                                       bP, dP, wfx, weB, xv0, out);
}
